// GraphSinkhornPooling_16492674417227
// MI455X (gfx1250) — matrix operand order repaired and checked by reading, not compiled, not run on hardware
//
#include <hip/hip_runtime.h>

// ---------------------------------------------------------------------------
// GraphSinkhornPooling on gfx1250 (MI455X)
//   B=128 graphs, NPG=512 nodes, K=256 atoms, D=128, eps=0.1, 100 iters
// Strategy:
//   * cost GEMM via v_wmma_f32_16x16x32_bf16 (A = hist rows, B = atom_d rows
//     exploiting symmetry -> all fragments are contiguous 128-bit loads)
//   * per-graph logK kept as f16 in LDS (512*256*2 = 256KB <= 320KB/WGP)
//   * all 100 Sinkhorn iterations run out of LDS, zero HBM traffic
// ---------------------------------------------------------------------------

#define NPG   512
#define KA    256
#define DIM   128
#define NB    128
#define INV_EPS 10.0f

typedef __attribute__((ext_vector_type(16))) __bf16 v16bf;
typedef __attribute__((ext_vector_type(8)))  __bf16 v8bf;
typedef __attribute__((ext_vector_type(8)))  float  v8f;

// ---- LDS layout (bytes) ----------------------------------------------------
#define OFF_LOGK 0          // 512*256 f16   = 262144
#define OFF_U    262144     // 512 f32       =   2048
#define OFF_V    264192     // 256 f32       =   1024
#define OFF_RM   265216     // 512 f32       =   2048
#define OFF_RS   267264     // 512 f32       =   2048
#define OFF_GH   269312     // 256 f32       =   1024
#define SMEM_BYTES 270336   // 264 KB (<= 320KB WGP LDS)

// ---------------------------------------------------------------------------
// Kernel 1: f32 -> bf16 conversion of node histograms
// ---------------------------------------------------------------------------
__global__ void k_hist_to_bf16(const float* __restrict__ in,
                               __bf16* __restrict__ out, int n) {
    int i = blockIdx.x * blockDim.x + threadIdx.x;
    int stride = gridDim.x * blockDim.x;
    for (; i < n; i += stride) out[i] = (__bf16)in[i];
}

// ---------------------------------------------------------------------------
// Kernel 2: atom_d[i][j] = ||cb_i - cb_j||^2 (symmetric), stored bf16
// ---------------------------------------------------------------------------
__global__ void k_atom_dist(const float* __restrict__ cb,
                            __bf16* __restrict__ atomd) {
    int i = blockIdx.x;      // 256 blocks
    int j = threadIdx.x;     // 256 threads
    const float* ri = cb + i * DIM;
    const float* rj = cb + j * DIM;
    float acc = 0.f;
    #pragma unroll 4
    for (int d = 0; d < DIM; ++d) {
        float df = ri[d] - rj[d];
        acc = fmaf(df, df, acc);
    }
    atomd[i * KA + j] = (__bf16)acc;
}

// ---------------------------------------------------------------------------
// helpers
// ---------------------------------------------------------------------------
__device__ __forceinline__ void lse_elem(float& m, float& s, float x) {
    float nm = fmaxf(m, x);
    s = s * __expf(m - nm) + __expf(x - nm);
    m = nm;
}
__device__ __forceinline__ void lse_merge(float& m, float& s, float om, float os) {
    float nm = fmaxf(m, om);
    s = s * __expf(m - nm) + os * __expf(om - nm);
    m = nm;
}

// ---------------------------------------------------------------------------
// Kernel 3: fused  cost-GEMM (WMMA) -> logK(f16, LDS) -> 100 Sinkhorn
//           iterations -> gh + emb.   One block per graph, 512 thr (16 waves)
// ---------------------------------------------------------------------------
__global__ void __launch_bounds__(512)
k_sinkhorn(const __bf16* __restrict__ histbf,
           const __bf16* __restrict__ atomd,
           const float*  __restrict__ cb,
           float* __restrict__ out) {
    extern __shared__ char smem[];
    _Float16* logK = (_Float16*)(smem + OFF_LOGK);
    float* u   = (float*)(smem + OFF_U);
    float* v   = (float*)(smem + OFF_V);
    float* rdm = (float*)(smem + OFF_RM);
    float* rds = (float*)(smem + OFF_RS);
    float* gh  = (float*)(smem + OFF_GH);

    const int b    = blockIdx.x;
    const int tid  = threadIdx.x;
    const int lane = tid & 31;
    const int wave = tid >> 5;          // 16 waves
    const int m16  = lane & 15;
    const int g    = lane >> 4;

    // ============ Phase A: cost GEMM via WMMA, fused logK -> LDS ============
    const __bf16* Abase = histbf + (size_t)b * NPG * KA;

    for (int sp = 0; sp < 2; ++sp) {
        const int row0 = (wave + sp * 16) * 16;     // 32 strips of 16 rows
        // Load the whole 16x256 A strip as 8 fragments (held in VGPRs,
        // reused across all 16 column tiles).
        v16bf A[8];
        #pragma unroll
        for (int ks = 0; ks < 8; ++ks) {
            const __bf16* p = Abase + (size_t)(row0 + m16) * KA + ks * 32 + 8 * g;
            v8bf lo = *(const v8bf*)p;          // K = ks*32 + 8g + 0..7
            v8bf hi = *(const v8bf*)(p + 16);   // K = ks*32 + 16 + 8g + 0..7
            #pragma unroll
            for (int e = 0; e < 8; ++e) { A[ks][e] = lo[e]; A[ks][e + 8] = hi[e]; }
        }
        for (int jt = 0; jt < 16; ++jt) {
            if (jt + 1 < 16)   // warm L2/L0 for next column panel
                __builtin_prefetch(atomd + (size_t)((jt + 1) * 16 + m16) * KA, 0, 1);
            v8f acc = {0.f, 0.f, 0.f, 0.f, 0.f, 0.f, 0.f, 0.f};
            #pragma unroll
            for (int ks = 0; ks < 8; ++ks) {
                // B fragment via symmetry: atomd[k][j] == atomd[j][k]
                // lane needs column n = m16, K = ks*32 + 8g + 0..7 and ks*32 + 16 + 8g + 0..7
                const __bf16* bp = atomd + (size_t)(jt * 16 + m16) * KA + ks * 32 + 8 * g;
                v8bf blo = *(const v8bf*)bp;
                v8bf bhi = *(const v8bf*)(bp + 16);
                v16bf Bf;
                #pragma unroll
                for (int e = 0; e < 8; ++e) { Bf[e] = blo[e]; Bf[e + 8] = bhi[e]; }
                acc = __builtin_amdgcn_wmma_f32_16x16x32_bf16(
                          false, A[ks], false, Bf, (short)0, acc, false, false);
            }
            // D layout: element r -> row = 8g + r, col = jt*16 + m16
            #pragma unroll
            for (int r = 0; r < 8; ++r) {
                float cost = acc[r];
                float lk = __logf(__expf(-INV_EPS * cost) + 1e-8f);
                int row = row0 + g * 8 + r;
                logK[row * KA + jt * 16 + m16] = (_Float16)lk;
            }
        }
    }

    // init duals
    u[tid] = 0.f;
    if (tid < KA) v[tid] = 0.f;
    __syncthreads();

    const float la = __logf(1.0f / (float)NPG + 1e-8f);
    const float lb = __logf(1.0f / (float)KA  + 1e-8f);

    // ============ Phase B: 100 Sinkhorn iterations, fully in LDS ============
    for (int it = 0; it < 100; ++it) {
        // ---- u update: wave per row, lane covers 8 contiguous columns ----
        for (int rp = 0; rp < 32; ++rp) {
            int row = rp * 16 + wave;
            const _Float16* lrow = logK + row * KA + lane * 8;
            float vm = -__builtin_inff(), vs = 0.f;
            #pragma unroll
            for (int q = 0; q < 8; ++q)
                lse_elem(vm, vs, (float)lrow[q] + v[lane * 8 + q]);
            #pragma unroll
            for (int off = 16; off > 0; off >>= 1) {
                float om = __shfl_xor(vm, off, 32);
                float os = __shfl_xor(vs, off, 32);
                lse_merge(vm, vs, om, os);
            }
            if (lane == 0) u[row] = la - (vm + __logf(vs));
        }
        __syncthreads();
        // ---- v update: 2 threads per column, 256 rows each ----
        {
            int col = tid & (KA - 1), hid = tid >> 8;
            float vm = -__builtin_inff(), vs = 0.f;
            const _Float16* lcol = logK + hid * 256 * KA + col;
            for (int q = 0; q < 256; ++q)
                lse_elem(vm, vs, (float)lcol[q * KA] + u[hid * 256 + q]);
            rdm[tid] = vm; rds[tid] = vs;
        }
        __syncthreads();
        if (tid < KA) {
            float m1 = rdm[tid], s1 = rds[tid];
            lse_merge(m1, s1, rdm[tid + KA], rds[tid + KA]);
            v[tid] = lb - (m1 + __logf(s1));
        }
        __syncthreads();
    }

    // ============ Phase C: gh = colsum(exp(u+v+logK)), normalize, emb ======
    {
        int col = tid & (KA - 1), hid = tid >> 8;
        float s = 0.f;
        const _Float16* lcol = logK + hid * 256 * KA + col;
        for (int q = 0; q < 256; ++q)
            s += __expf((float)lcol[q * KA] + u[hid * 256 + q]);
        rds[tid] = s;
    }
    __syncthreads();
    if (tid < KA) gh[tid] = __expf(v[tid]) * (rds[tid] + rds[tid + KA]);
    __syncthreads();
    if (tid < KA) rds[tid] = gh[tid];
    __syncthreads();
    for (int st = 128; st > 0; st >>= 1) {
        if (tid < st) rds[tid] += rds[tid + st];
        __syncthreads();
    }
    float total = rds[0];
    if (tid < KA) {
        float gn = gh[tid] / total;
        out[(size_t)b * KA + tid] = gn;   // gh output [B,K]
        gh[tid] = gn;
    }
    __syncthreads();
    if (tid < DIM) {
        float acc = 0.f;
        #pragma unroll 4
        for (int k = 0; k < KA; ++k)
            acc = fmaf(gh[k], cb[k * DIM + tid], acc);
        out[NB * KA + (size_t)b * DIM + tid] = acc;  // emb output [B,D]
    }
}

// ---------------------------------------------------------------------------
extern "C" void kernel_launch(void* const* d_in, const int* in_sizes, int n_in,
                              void* d_out, int out_size, void* d_ws, size_t ws_size,
                              hipStream_t stream) {
    const float* hist = (const float*)d_in[0];   // [B*NPG, K] f32
    // d_in[1] = batch_idx (equal sorted segments -> unused)
    const float* cb   = (const float*)d_in[2];   // [K, D] f32
    float* out = (float*)d_out;                  // gh (B*K) ++ emb (B*D)

    // workspace: atom_d bf16 (128KB) then hist bf16 (32MB)
    __bf16* atomd  = (__bf16*)d_ws;
    __bf16* histbf = (__bf16*)((char*)d_ws + (size_t)KA * KA * sizeof(__bf16));

    const int nh = in_sizes[0];                  // B*NPG*K
    k_hist_to_bf16<<<4096, 256, 0, stream>>>(hist, histbf, nh);
    k_atom_dist  <<<KA, KA, 0, stream>>>(cb, atomd);
    k_sinkhorn   <<<NB, 512, SMEM_BYTES, stream>>>(histbf, atomd, cb, out);
}
